// FusionRetrModelBase_58076547776826
// MI455X (gfx1250) — matrix operand order repaired and checked by reading, not compiled, not run on hardware
//
#include <hip/hip_runtime.h>
#include <hip/hip_bf16.h>

// ---------------------------------------------------------------------------
// FusionRetrModelBase for MI455X (gfx1250, wave32, WMMA).
// Compute-bound (~242 GFLOP vs ~100 MB live data @ 23.3 TB/s) => run the two
// big GEMMs on v_wmma_f32_16x16x32_bf16 (bf16 in, fp32 accumulate), with
// double-buffered GLOBAL_LOAD_ASYNC_TO_LDS_B128 staging (ASYNCcnt pipeline).
// ---------------------------------------------------------------------------

typedef __bf16 bf16_t;
typedef __attribute__((ext_vector_type(16))) __bf16 v16bf;
typedef __attribute__((ext_vector_type(8)))  __bf16 v8bf;
typedef __attribute__((ext_vector_type(8)))  float  v8f;
typedef __attribute__((ext_vector_type(4)))  int    v4i;

#if __has_builtin(__builtin_amdgcn_global_load_async_to_lds_b128) && \
    __has_builtin(__builtin_amdgcn_s_wait_asynccnt)
#define USE_ASYNC 1
#else
#define USE_ASYNC 0
#endif

#if USE_ASYNC
// builtin signature (from hipcc diagnostic): (v4i addrspace(1)*, v4i addrspace(3)*, Imm, Imm)
typedef __attribute__((address_space(1))) v4i* g4ip;
typedef __attribute__((address_space(3))) v4i* l4ip;
__device__ __forceinline__ g4ip to_global(const void* p) {
  return (g4ip)(unsigned long long)p;
}
__device__ __forceinline__ l4ip to_lds(void* p) {
  // generic LDS address: addr[31:0] is the LDS offset (ISA 10.2 aperture rules)
  return (l4ip)(unsigned int)(unsigned long long)p;
}
#endif

#define B_    4
#define L_    2
#define P_    100
#define TPP_  64
#define D_    768
#define G_    10
#define T_    (P_ * TPP_)          // 6400
#define M_    (B_ * T_)            // 25600 rows
#define K1_   (3 * D_)             // 2304
#define N1_   (2 * D_)             // 1536 ( [W_p | W_t] fused )
#define K2_   (2 * D_)             // 1536
#define N2_   (D_)                 // 768

// workspace layout (bytes, all offsets 256B aligned)
#define OFF_X      0ull                                   // 25600*2304 bf16
#define OFF_FEAT   (OFF_X    + (size_t)M_ * K1_ * 2)      // 25600*1536 bf16
#define OFF_TS     (OFF_FEAT + (size_t)M_ * N1_ * 2)      // 25600* 768 bf16
#define OFF_WB1    (OFF_TS   + (size_t)M_ * D_  * 2)      // 1536*2304 bf16
#define OFF_WB2    (OFF_WB1  + (size_t)N1_ * K1_ * 2)     //  768*1536 bf16
#define OFF_BIAS1  (OFF_WB2  + (size_t)N2_ * K2_ * 2)     // 1536 f32
#define OFF_SCORES (OFF_BIAS1 + (size_t)N1_ * 4)          // 25600 f32

// ---------------------------------------------------------------------------
// prep: X = bf16([a | q | a*q]) for the l = L-1 slice, row-major 25600 x 2304
// ---------------------------------------------------------------------------
__global__ void k_build_x(const float* __restrict__ ans,
                          const float* __restrict__ qps,
                          bf16_t* __restrict__ X) {
  int r = blockIdx.x;                       // 0..M_-1
  int b = r / T_, t = r - b * T_;
  const float* arow = ans + (size_t)(b * L_ + (L_ - 1)) * D_;
  const float* qrow = qps + ((size_t)(b * L_ + (L_ - 1)) * T_ + t) * D_;
  bf16_t* xrow = X + (size_t)r * K1_;
  for (int k = threadIdx.x; k < K1_; k += blockDim.x) {
    int region = k / D_;
    int d = k - region * D_;
    float a = arow[d], q = qrow[d];
    float v = (region == 0) ? a : (region == 1) ? q : a * q;
    xrow[k] = (bf16_t)v;
  }
}

// ---------------------------------------------------------------------------
// prep: weights -> bf16, transposed to [N][K] so GEMM B tiles are contiguous.
// WB1t[n][k] = (n<768 ? W_p : W_t)[k][n%768];  WB2t[n][k] = W1[k][n]
// bias1 = [b_p | b_t]
// ---------------------------------------------------------------------------
__global__ void k_build_w(const float* __restrict__ Wp, const float* __restrict__ Wt,
                          const float* __restrict__ bp, const float* __restrict__ bt,
                          const float* __restrict__ W1,
                          bf16_t* __restrict__ WB1t, bf16_t* __restrict__ WB2t,
                          float* __restrict__ bias1) {
  long i0 = (long)blockIdx.x * blockDim.x + threadIdx.x;
  long stride = (long)gridDim.x * blockDim.x;
  for (long i = i0; i < (long)N1_ * K1_; i += stride) {
    int n = (int)(i / K1_), k = (int)(i - (long)n * K1_);
    float v = (n < D_) ? Wp[(size_t)k * D_ + n] : Wt[(size_t)k * D_ + (n - D_)];
    WB1t[i] = (bf16_t)v;
  }
  for (long i = i0; i < (long)N2_ * K2_; i += stride) {
    int n = (int)(i / K2_), k = (int)(i - (long)n * K2_);
    WB2t[i] = (bf16_t)W1[(size_t)k * D_ + n];
  }
  for (long i = i0; i < N1_; i += stride)
    bias1[i] = (i < D_) ? bp[i] : bt[i - D_];
}

// ---------------------------------------------------------------------------
// stage one 128x64 bf16 slice of A and of Bt into LDS.
// 256 threads x 4 chunks of 16B each per matrix => 8 async ops per wave.
// ---------------------------------------------------------------------------
template <int K>
__device__ __forceinline__ void stage(const bf16_t* __restrict__ A,
                                      const bf16_t* __restrict__ Bt,
                                      int blockM, int blockN, int k0,
                                      bf16_t* As, bf16_t* Bs) {
  const int tid = threadIdx.x;
#if USE_ASYNC
#pragma unroll
  for (int h = 0; h < 4; ++h) {
    int i = tid + h * 256;            // 0..1023
    int r = i >> 3;                   // row 0..127
    int c = (i & 7) * 8;              // col (bf16 elems), 16B chunks
    __builtin_amdgcn_global_load_async_to_lds_b128(
        to_global(&A[(size_t)(blockM + r) * K + k0 + c]),
        to_lds(&As[r * 64 + c]), 0, 0);
    __builtin_amdgcn_global_load_async_to_lds_b128(
        to_global(&Bt[(size_t)(blockN + r) * K + k0 + c]),
        to_lds(&Bs[r * 64 + c]), 0, 0);
  }
#else
  // fallback: two-phase (all loads issued, then all stores) to pipeline loadcnt
  v8bf ra[4], rb[4];
#pragma unroll
  for (int h = 0; h < 4; ++h) {
    int i = tid + h * 256;
    int r = i >> 3, c = (i & 7) * 8;
    ra[h] = *(const v8bf*)&A [(size_t)(blockM + r) * K + k0 + c];
    rb[h] = *(const v8bf*)&Bt[(size_t)(blockN + r) * K + k0 + c];
  }
#pragma unroll
  for (int h = 0; h < 4; ++h) {
    int i = tid + h * 256;
    int r = i >> 3, c = (i & 7) * 8;
    *(v8bf*)&As[r * 64 + c] = ra[h];
    *(v8bf*)&Bs[r * 64 + c] = rb[h];
  }
#endif
}

// ---------------------------------------------------------------------------
// WMMA mainloop: C(128x128) += A(128xK) * Bt(128xK)^T, bf16 in / f32 acc.
// 8 waves as 4(M) x 2(N); wave tile 32x64 = 2x4 wmma tiles; K sliced by 64
// with double-buffered LDS and async staging overlapped with compute.
// Fragment layouts per CDNA5 ISA 7.12.2 (16-bit A 16x32, B 32x16, f32 C).
// ---------------------------------------------------------------------------
template <int K>
__device__ __forceinline__ void gemm_tile(const bf16_t* __restrict__ A,
                                          const bf16_t* __restrict__ Bt,
                                          int blockM, int blockN,
                                          bf16_t* As0, bf16_t* Bs0,
                                          bf16_t* As1, bf16_t* Bs1,
                                          v8f (&acc)[2][4]) {
  const int lane  = threadIdx.x & 31;
  const int wave  = threadIdx.x >> 5;
  const int wm    = wave >> 1;      // 0..3
  const int wn    = wave & 1;       // 0..1
  const int lhalf = lane >> 4;      // 0/1
  const int lmod  = lane & 15;
  constexpr int NIT = K / 64;

  stage<K>(A, Bt, blockM, blockN, 0, As0, Bs0);

#pragma unroll 2
  for (int it = 0; it < NIT; ++it) {
    bf16_t* Ac = (it & 1) ? As1 : As0;
    bf16_t* Bc = (it & 1) ? Bs1 : Bs0;
    if (it + 1 < NIT) {
      stage<K>(A, Bt, blockM, blockN, (it + 1) * 64,
               (it & 1) ? As0 : As1, (it & 1) ? Bs0 : Bs1);
#if USE_ASYNC
      __builtin_amdgcn_s_wait_asynccnt(8);   // in-order: current buffer's 8 done
#endif
    } else {
#if USE_ASYNC
      __builtin_amdgcn_s_wait_asynccnt(0);
#endif
    }
    __syncthreads();

#pragma unroll
    for (int ks = 0; ks < 64; ks += 32) {
      v16bf afr[2], bfr[4];
#pragma unroll
      for (int mt = 0; mt < 2; ++mt) {       // A: lane -> M=lmod, K half by lane[4]
        union { v16bf v; v8bf h[2]; } u;
        int ar = wm * 32 + mt * 16 + lmod;
        u.h[0] = *(const v8bf*)&Ac[ar * 64 + ks + lhalf * 8];
        u.h[1] = *(const v8bf*)&Ac[ar * 64 + ks + lhalf * 8 + 16];
        afr[mt] = u.v;
      }
#pragma unroll
      for (int nt = 0; nt < 4; ++nt) {       // B: lane -> N=lmod, K half by lane[4]
        union { v16bf v; v8bf h[2]; } u;
        int bn = wn * 64 + nt * 16 + lmod;
        u.h[0] = *(const v8bf*)&Bc[bn * 64 + ks + lhalf * 8];
        u.h[1] = *(const v8bf*)&Bc[bn * 64 + ks + lhalf * 8 + 16];
        bfr[nt] = u.v;
      }
#pragma unroll
      for (int mt = 0; mt < 2; ++mt)
#pragma unroll
        for (int nt = 0; nt < 4; ++nt)
          acc[mt][nt] = __builtin_amdgcn_wmma_f32_16x16x32_bf16(
              false, afr[mt], false, bfr[nt], (short)0, acc[mt][nt], false, false);
    }
    __syncthreads();
  }
}

// ---------------------------------------------------------------------------
// GEMM1: [ps|ts] = X @ [W_p|W_t] + [b_p|b_t].
// cols < 768 -> ps into FEAT left half (bf16); cols >= 768 -> ts scratch (bf16)
// ---------------------------------------------------------------------------
__global__ void __launch_bounds__(256) k_gemm1(const bf16_t* __restrict__ X,
                                               const bf16_t* __restrict__ WB1t,
                                               const float* __restrict__ bias1,
                                               bf16_t* __restrict__ FEAT,
                                               bf16_t* __restrict__ TS) {
  __shared__ bf16_t As0[128 * 64], Bs0[128 * 64];
  __shared__ bf16_t As1[128 * 64], Bs1[128 * 64];
  v8f acc[2][4] = {};
  const int blockM = blockIdx.x * 128;
  const int blockN = blockIdx.y * 128;
  gemm_tile<K1_>(X, WB1t, blockM, blockN, As0, Bs0, As1, Bs1, acc);

  const int lane = threadIdx.x & 31, wave = threadIdx.x >> 5;
  const int wm = wave >> 1, wn = wave & 1;
  const int lhalf = lane >> 4, lmod = lane & 15;
#pragma unroll
  for (int mt = 0; mt < 2; ++mt)
#pragma unroll
    for (int nt = 0; nt < 4; ++nt)
#pragma unroll
      for (int j = 0; j < 8; ++j) {           // C: VGPR j -> M = j + 8*lane[4]
        int row = blockM + wm * 32 + mt * 16 + lhalf * 8 + j;
        int col = blockN + wn * 64 + nt * 16 + lmod;
        float c = acc[mt][nt][j] + bias1[col];
        if (col < D_) FEAT[(size_t)row * N1_ + col] = (bf16_t)c;
        else          TS  [(size_t)row * D_ + (col - D_)] = (bf16_t)c;
      }
}

// ---------------------------------------------------------------------------
// segment-max of ts over passages (per batch, per (tpp,d)); gather by table id;
// write into FEAT right half. One block per (b, tpp); 10 group maxima live in
// registers via unrolled selects (no scratch spills).
// ---------------------------------------------------------------------------
__global__ void __launch_bounds__(256) k_aggr(const bf16_t* __restrict__ TS,
                                              const int* __restrict__ table_ids,
                                              bf16_t* __restrict__ FEAT) {
  const int b = blockIdx.x / TPP_, tpp = blockIdx.x % TPP_;
  __shared__ int stid[P_];
  if (threadIdx.x < P_) stid[threadIdx.x] = table_ids[b * P_ + threadIdx.x];
  __syncthreads();

  float gmax[3][G_];
#pragma unroll
  for (int dd = 0; dd < 3; ++dd)
#pragma unroll
    for (int g = 0; g < G_; ++g) gmax[dd][g] = -3.4e38f;

  for (int p = 0; p < P_; ++p) {
    int sg = stid[p];
    size_t rbase = (size_t)((b * P_ + p) * TPP_ + tpp) * D_;
#pragma unroll
    for (int dd = 0; dd < 3; ++dd) {
      float v = (float)TS[rbase + threadIdx.x + dd * 256];
#pragma unroll
      for (int g = 0; g < G_; ++g)
        gmax[dd][g] = (sg == g) ? fmaxf(gmax[dd][g], v) : gmax[dd][g];
    }
  }
  for (int p = 0; p < P_; ++p) {
    int sg = stid[p];
    size_t rbase = (size_t)((b * P_ + p) * TPP_ + tpp) * N1_ + D_;
#pragma unroll
    for (int dd = 0; dd < 3; ++dd) {
      float v = 0.f;
#pragma unroll
      for (int g = 0; g < G_; ++g) v = (sg == g) ? gmax[dd][g] : v;
      FEAT[rbase + threadIdx.x + dd * 256] = (bf16_t)v;
    }
  }
}

// ---------------------------------------------------------------------------
// GEMM2 fused epilogue: h = relu(FEAT @ W1 + b1); scores += h . W2
// (per-row partial over this block's 128 cols -> shfl reduce -> atomicAdd)
// ---------------------------------------------------------------------------
__global__ void __launch_bounds__(256) k_gemm2(const bf16_t* __restrict__ FEAT,
                                               const bf16_t* __restrict__ WB2t,
                                               const float* __restrict__ bias,
                                               const float* __restrict__ W2,
                                               float* __restrict__ scores) {
  __shared__ bf16_t As0[128 * 64], Bs0[128 * 64];
  __shared__ bf16_t As1[128 * 64], Bs1[128 * 64];
  v8f acc[2][4] = {};
  const int blockM = blockIdx.x * 128;
  const int blockN = blockIdx.y * 128;
  gemm_tile<K2_>(FEAT, WB2t, blockM, blockN, As0, Bs0, As1, Bs1, acc);

  const int lane = threadIdx.x & 31, wave = threadIdx.x >> 5;
  const int wm = wave >> 1, wn = wave & 1;
  const int lhalf = lane >> 4, lmod = lane & 15;

  float rsum[2][8] = {};
#pragma unroll
  for (int mt = 0; mt < 2; ++mt)
#pragma unroll
    for (int nt = 0; nt < 4; ++nt)
#pragma unroll
      for (int j = 0; j < 8; ++j) {
        int col = blockN + wn * 64 + nt * 16 + lmod;
        float c = fmaxf(acc[mt][nt][j] + bias[col], 0.f);
        rsum[mt][j] += c * W2[col];
      }
#pragma unroll
  for (int mt = 0; mt < 2; ++mt)
#pragma unroll
    for (int j = 0; j < 8; ++j) {
      float v = rsum[mt][j];                  // reduce over the 16 N-lanes
      v += __shfl_xor(v, 1, 32);
      v += __shfl_xor(v, 2, 32);
      v += __shfl_xor(v, 4, 32);
      v += __shfl_xor(v, 8, 32);
      if (lmod == 0) {
        int row = blockM + wm * 32 + mt * 16 + lhalf * 8 + j;
        atomicAdd(&scores[row], v);
      }
    }
}

// ---------------------------------------------------------------------------
// finalize: out[b,p] = sum_t (scores + b2) * mask
// ---------------------------------------------------------------------------
__global__ void k_out(const float* __restrict__ scores,
                      const float* __restrict__ mask,
                      const float* __restrict__ b2,
                      float* __restrict__ out) {
  int i = blockIdx.x * blockDim.x + threadIdx.x;
  if (i >= B_ * P_) return;
  float bb = b2[0];
  const float* sc = scores + (size_t)i * TPP_;
  const float* mk = mask   + (size_t)i * TPP_;
  float s = 0.f;
#pragma unroll 8
  for (int t = 0; t < TPP_; ++t) s += (sc[t] + bb) * mk[t];
  out[i] = s;
}

// ---------------------------------------------------------------------------
extern "C" void kernel_launch(void* const* d_in, const int* in_sizes, int n_in,
                              void* d_out, int out_size, void* d_ws, size_t ws_size,
                              hipStream_t stream) {
  (void)in_sizes; (void)n_in; (void)out_size; (void)ws_size;
  const float* ans  = (const float*)d_in[0];
  const float* qps  = (const float*)d_in[1];
  const float* mask = (const float*)d_in[2];
  const int*   tids = (const int*)  d_in[3];
  /* d_in[4] fusion_scores: unused by the reference */
  const float* Wp = (const float*)d_in[5];
  const float* bp = (const float*)d_in[6];
  const float* Wt = (const float*)d_in[7];
  const float* bt = (const float*)d_in[8];
  const float* W1 = (const float*)d_in[9];
  const float* b1 = (const float*)d_in[10];
  const float* W2 = (const float*)d_in[11];
  const float* b2 = (const float*)d_in[12];
  float* out = (float*)d_out;

  char* ws = (char*)d_ws;
  bf16_t* X      = (bf16_t*)(ws + OFF_X);
  bf16_t* FEAT   = (bf16_t*)(ws + OFF_FEAT);
  bf16_t* TS     = (bf16_t*)(ws + OFF_TS);
  bf16_t* WB1t   = (bf16_t*)(ws + OFF_WB1);
  bf16_t* WB2t   = (bf16_t*)(ws + OFF_WB2);
  float*  bias1  = (float*) (ws + OFF_BIAS1);
  float*  scores = (float*) (ws + OFF_SCORES);

  (void)hipMemsetAsync(scores, 0, (size_t)M_ * sizeof(float), stream);

  k_build_x<<<M_, 256, 0, stream>>>(ans, qps, X);
  k_build_w<<<2048, 256, 0, stream>>>(Wp, Wt, bp, bt, W1, WB1t, WB2t, bias1);

  k_gemm1<<<dim3(M_ / 128, N1_ / 128), 256, 0, stream>>>(X, WB1t, bias1, FEAT, TS);
  k_aggr <<<B_ * TPP_, 256, 0, stream>>>(TS, tids, FEAT);
  k_gemm2<<<dim3(M_ / 128, N2_ / 128), 256, 0, stream>>>(FEAT, WB2t, b1, W2, scores);
  k_out  <<<(B_ * P_ + 255) / 256, 256, 0, stream>>>(scores, mask, b2, out);
}
